// GATAutoencoder_24326694764710
// MI455X (gfx1250) — matrix operand order repaired and checked by reading, not compiled, not run on hardware
//
#include <hip/hip_runtime.h>
#include <math.h>

// ---------------------------------------------------------------------------
// GAT autoencoder for MI455X (gfx1250, wave32).
// GEMMs run on v_wmma_f32_16x16x32_f16 (f16 in, f32 accum).
// Edge gather/scatter kernels rely on the 192MB L2: all node-feature buffers
// (<= 51.2MB) fit in L2, so random src/dst traffic never reaches HBM.
// ---------------------------------------------------------------------------

#define N_NODES 50000
#define N_EDGES 800000
#define ETOT    (N_EDGES + N_NODES)

typedef __attribute__((ext_vector_type(16))) _Float16 v16h;
typedef __attribute__((ext_vector_type(8)))  _Float16 v8h;
typedef __attribute__((ext_vector_type(8)))  float    v8f;

static inline int ceil_div_i(int a, int b) { return (a + b - 1) / b; }

// ---------------- float atomic max (sign-aware int trick) -------------------
__device__ __forceinline__ void atomic_max_f32(float* addr, float v) {
  if (v >= 0.0f)
    atomicMax((int*)addr, __float_as_int(v));
  else
    atomicMin((unsigned int*)addr, __float_as_uint(v));
}

// self-loops appended after the E real edges
__device__ __forceinline__ void edge_sd(const int* __restrict__ ei, int e, int& s, int& d) {
  if (e < N_EDGES) { s = ei[e]; d = ei[N_EDGES + e]; }
  else             { s = e - N_EDGES; d = s; }
}

// ---------------------------- small utility kernels -------------------------
__global__ void cvt_f16_kernel(const float* __restrict__ src, _Float16* __restrict__ dst, int n) {
  int i = blockIdx.x * blockDim.x + threadIdx.x;
  if (i < n) dst[i] = (_Float16)src[i];
}

__global__ void zero_kernel(float* __restrict__ p, int n) {
  int i = blockIdx.x * blockDim.x + threadIdx.x;
  if (i < n) p[i] = 0.0f;
}

__global__ void init_attn_kernel(float* __restrict__ emax, float* __restrict__ denom, int n) {
  int i = blockIdx.x * blockDim.x + threadIdx.x;
  if (i < n) { emax[i] = -__builtin_inff(); denom[i] = 0.0f; }
}

// out[n*F + f] = bias[f]   (F is a power of two: 128 or 256)
__global__ void init_bias_kernel(float* __restrict__ out, const float* __restrict__ bias, int F) {
  int i = blockIdx.x * blockDim.x + threadIdx.x;
  if (i < N_NODES * F) out[i] = bias[i & (F - 1)];
}

__global__ void relu_kernel(float* __restrict__ x, int n) {
  int i = blockIdx.x * blockDim.x + threadIdx.x;
  if (i < n) x[i] = fmaxf(x[i], 0.0f);
}

// ---------------------- weight packing for WMMA B-fragments -----------------
// Layout per (tileN, kblock): 32 lanes x 16 halves.
// Lane l supplies column n = tileN*16 + (l&15); element j holds
// B[kblock*32 + 16*(l>>4) + j][n]  (contiguous k order, same as the A fragment).
__global__ void pack_w_kernel(const float* __restrict__ W, _Float16* __restrict__ Bp,
                              int K, int Nc) {
  int idx = blockIdx.x * blockDim.x + threadIdx.x;
  int KB = K >> 5;
  int total = (Nc >> 4) * KB * 512;
  if (idx >= total) return;
  int j    = idx & 15;
  int lane = (idx >> 4) & 31;
  int blk  = idx >> 9;           // tn*KB + kb
  int kb   = blk % KB;
  int tn   = blk / KB;
  int k = kb * 32 + ((lane >> 4) << 4) + j;
  int n = tn * 16 + (lane & 15);
  Bp[idx] = (_Float16)W[k * Nc + n];
}

// ------------------------------- WMMA GEMM ----------------------------------
// C[M,Nc] = A[M,K] @ B[K,Nc] (+bias). A: f16 row-major; B: packed f16; C: f32.
// One 16x16 tile per wave, 4 waves per block. M % 16 == 0, K % 32 == 0,
// Nc % 16 == 0 and gridDim.y*4 tiles cover Nc exactly -> EXEC all ones.
__global__ void wmma_gemm_kernel(const _Float16* __restrict__ A,
                                 const _Float16* __restrict__ Bp,
                                 const float* __restrict__ bias,
                                 float* __restrict__ C,
                                 int M, int K, int Nc) {
  int lane  = threadIdx.x & 31;
  int wave  = threadIdx.x >> 5;
  int tileM = blockIdx.x;
  int tileN = blockIdx.y * 4 + wave;
  if (tileN * 16 >= Nc) return;          // wave-uniform
  int hsel = lane >> 4;
  int m = tileM * 16 + (lane & 15);

  // A fragment (16-bit A 16x32): lane half h: elems 0..15 = K=kk+16h..+15
  // (matches packed B) -> two contiguous 16B loads.
  const _Float16* arow = A + (size_t)m * K + hsel * 16;
  const _Float16* bptr = Bp + ((size_t)tileN * (K >> 5)) * 512 + lane * 16;

  v8f acc = {};
  for (int kk = 0; kk < K; kk += 32) {
    v8h alo = *(const v8h*)(arow + kk);
    v8h ahi = *(const v8h*)(arow + kk + 8);
    v16h af;
#pragma unroll
    for (int j = 0; j < 8; j++) { af[j] = alo[j]; af[j + 8] = ahi[j]; }
    v16h bf = *(const v16h*)bptr;
    bptr += 512;
    acc = __builtin_amdgcn_wmma_f32_16x16x32_f16(
        /*neg_a=*/false, af, /*neg_b=*/false, bf,
        /*c_mod=*/(short)0, acc, /*reuse_a=*/false, /*reuse_b=*/false);
  }

  int col = tileN * 16 + (lane & 15);
  float bv = bias ? bias[col] : 0.0f;
#pragma unroll
  for (int r = 0; r < 8; r++) {          // C/D: VGPR r -> M = r + 8*hsel
    int row = tileM * 16 + r + hsel * 8;
    C[(size_t)row * Nc + col] = acc[r] + bv;
  }
}

// --------------------------- attention logits -------------------------------
// als[n,h] = sum_c h[n,h,c]*a_src[h,c] ; ald likewise.
__global__ void attn_logits_kernel(const float* __restrict__ h,
                                   const float* __restrict__ a_s,
                                   const float* __restrict__ a_d,
                                   float* __restrict__ als, float* __restrict__ ald,
                                   int H, int C) {
  int n = blockIdx.x * blockDim.x + threadIdx.x;
  if (n >= N_NODES) return;
  const float* hr = h + (size_t)n * H * C;
  for (int hh = 0; hh < H; hh++) {
    float s = 0.0f, d = 0.0f;
    for (int c = 0; c < C; c++) {
      float v = hr[hh * C + c];
      s += v * a_s[hh * C + c];
      d += v * a_d[hh * C + c];
    }
    als[n * H + hh] = s;
    ald[n * H + hh] = d;
  }
}

// ----------------------------- edge passes ----------------------------------
__global__ void edge_max_kernel(const int* __restrict__ ei, const float* __restrict__ als,
                                const float* __restrict__ ald, float* __restrict__ emax, int H) {
  int idx = blockIdx.x * blockDim.x + threadIdx.x;
  if (idx >= ETOT * H) return;
  int e = idx / H, hh = idx % H;
  int s, d; edge_sd(ei, e, s, d);
  float xv = als[s * H + hh] + ald[d * H + hh];
  xv = xv > 0.0f ? xv : 0.2f * xv;       // leaky_relu(0.2)
  atomic_max_f32(&emax[d * H + hh], xv);
}

__global__ void edge_exp_kernel(const int* __restrict__ ei, const float* __restrict__ als,
                                const float* __restrict__ ald, const float* __restrict__ emax,
                                float* __restrict__ wbuf, float* __restrict__ denom, int H) {
  int idx = blockIdx.x * blockDim.x + threadIdx.x;
  if (idx >= ETOT * H) return;
  int e = idx / H, hh = idx % H;
  int s, d; edge_sd(ei, e, s, d);
  float xv = als[s * H + hh] + ald[d * H + hh];
  xv = xv > 0.0f ? xv : 0.2f * xv;
  float m = emax[d * H + hh];
  if (!(m > -1e38f && m < 1e38f)) m = 0.0f;   // where(isfinite(e_max), e_max, 0)
  float w = __expf(xv - m);
  wbuf[idx] = w;
  atomicAdd(&denom[d * H + hh], w);
}

// block per edge, thread per feature: coalesced gather of h[src] row,
// coalesced atomic scatter into out[dst] row (all resident in L2).
__global__ void edge_aggr_kernel(const int* __restrict__ ei, const float* __restrict__ h,
                                 const float* __restrict__ wbuf, const float* __restrict__ denom,
                                 float* __restrict__ out, int H, int C) {
  int e = blockIdx.x;
  int f = threadIdx.x;                   // blockDim.x == H*C
  int F = H * C;
  int s, d; edge_sd(ei, e, s, d);
  int hh = f / C;
  float alpha = wbuf[e * H + hh] / (denom[d * H + hh] + 1e-16f);
  atomicAdd(&out[(size_t)d * F + f], h[(size_t)s * F + f] * alpha);
}

// ------------------------- fused ReLU + BatchNorm ---------------------------
__global__ void bn_partial_kernel(const float* __restrict__ x,
                                  float* __restrict__ musum, float* __restrict__ m2sum, int F) {
  int f = threadIdx.x;                   // blockDim.x == F
  float s = 0.0f, s2 = 0.0f;
  for (int n = blockIdx.x; n < N_NODES; n += gridDim.x) {
    float v = fmaxf(x[(size_t)n * F + f], 0.0f);   // relu before BN
    s += v; s2 += v * v;
  }
  atomicAdd(&musum[f], s);
  atomicAdd(&m2sum[f], s2);
}

__global__ void bn_final_kernel(const float* __restrict__ musum, const float* __restrict__ m2sum,
                                float* __restrict__ mu, float* __restrict__ rstd) {
  int f = threadIdx.x;
  float m = musum[f] / (float)N_NODES;
  float v = m2sum[f] / (float)N_NODES - m * m;     // population variance (ddof=0)
  mu[f] = m;
  rstd[f] = rsqrtf(v + 1e-5f);
}

__global__ void bn_apply_kernel(float* __restrict__ x, const float* __restrict__ mu,
                                const float* __restrict__ rstd, const float* __restrict__ g,
                                const float* __restrict__ be, int F) {
  int i = blockIdx.x * blockDim.x + threadIdx.x;
  if (i >= N_NODES * F) return;
  int f = i & (F - 1);
  float v = fmaxf(x[i], 0.0f);
  x[i] = (v - mu[f]) * rstd[f] * g[f] + be[f];
}

// ---------------------- inner-product decoder (L=64) ------------------------
__global__ void adj_decode_kernel(const int* __restrict__ ei, const float* __restrict__ z,
                                  float* __restrict__ out) {
  int e    = blockIdx.x * (blockDim.x >> 5) + (threadIdx.x >> 5);
  int lane = threadIdx.x & 31;
  if (e >= N_EDGES) return;
  int s = ei[e], d = ei[N_EDGES + e];
  const float* zs = z + (size_t)s * 64;
  const float* zd = z + (size_t)d * 64;
  float acc = zs[lane] * zd[lane] + zs[lane + 32] * zd[lane + 32];
  for (int off = 16; off > 0; off >>= 1) acc += __shfl_down(acc, off, 32);
  if (lane == 0) out[e] = 1.0f / (1.0f + __expf(-acc));
}

// ---------------------------------------------------------------------------
// Host-side orchestration
// ---------------------------------------------------------------------------
struct Scratch {
  float* B0;        // N*256 f32: GEMM output h_pre (and later z)
  float* B1;        // N*256 f32: activation in/out
  _Float16* A16;    // N*256 f16: activation converted for WMMA
  float *als, *ald, *emax, *denom;
  float *musum, *m2sum, *mu, *rstd;
  float* wbuf;      // ETOT*2 per-edge softmax numerators
  _Float16* Wp;     // packed weights, <= 32768 halves
};

static void launch_gemm(const _Float16* A16, const float* W, _Float16* Wp,
                        const float* bias, float* Cout, int K, int Nc, hipStream_t s) {
  int packN = (Nc / 16) * (K / 32) * 512;
  pack_w_kernel<<<ceil_div_i(packN, 256), 256, 0, s>>>(W, Wp, K, Nc);
  dim3 grid(N_NODES / 16, Nc / 64);
  wmma_gemm_kernel<<<grid, 128, 0, s>>>(A16, Wp, bias, Cout, N_NODES, K, Nc);
}

static void run_gat(const float* act_in, int Fin, int H, int C,
                    const float* W, const float* a_s, const float* a_d, const float* bias,
                    const int* ei, const Scratch& S, hipStream_t s) {
  int F = H * C;
  int nact = N_NODES * Fin;
  // act_in may alias S.B1: consumed entirely by this conversion before B1 is rewritten.
  cvt_f16_kernel<<<ceil_div_i(nact, 256), 256, 0, s>>>(act_in, S.A16, nact);
  launch_gemm(S.A16, W, S.Wp, nullptr, S.B0, Fin, F, s);   // h_pre = act @ W
  attn_logits_kernel<<<ceil_div_i(N_NODES, 256), 256, 0, s>>>(S.B0, a_s, a_d, S.als, S.ald, H, C);
  init_attn_kernel<<<ceil_div_i(N_NODES * H, 256), 256, 0, s>>>(S.emax, S.denom, N_NODES * H);
  edge_max_kernel<<<ceil_div_i(ETOT * H, 256), 256, 0, s>>>(ei, S.als, S.ald, S.emax, H);
  edge_exp_kernel<<<ceil_div_i(ETOT * H, 256), 256, 0, s>>>(ei, S.als, S.ald, S.emax, S.wbuf, S.denom, H);
  init_bias_kernel<<<ceil_div_i(N_NODES * F, 256), 256, 0, s>>>(S.B1, bias, F);
  edge_aggr_kernel<<<ETOT, F, 0, s>>>(ei, S.B0, S.wbuf, S.denom, S.B1, H, C);
}

static void run_bn(float* x, const float* g, const float* be, int F, const Scratch& S, hipStream_t s) {
  zero_kernel<<<2, 256, 0, s>>>(S.musum, 512);             // musum + m2sum contiguous
  bn_partial_kernel<<<64, F, 0, s>>>(x, S.musum, S.m2sum, F);
  bn_final_kernel<<<1, F, 0, s>>>(S.musum, S.m2sum, S.mu, S.rstd);
  bn_apply_kernel<<<ceil_div_i(N_NODES * F, 256), 256, 0, s>>>(x, S.mu, S.rstd, g, be, F);
}

extern "C" void kernel_launch(void* const* d_in, const int* in_sizes, int n_in,
                              void* d_out, int out_size, void* d_ws, size_t ws_size,
                              hipStream_t stream) {
  const float* x   = (const float*)d_in[0];
  const int*   ei  = (const int*)d_in[1];
  const float* W1  = (const float*)d_in[2];
  const float* as1 = (const float*)d_in[3];
  const float* ad1 = (const float*)d_in[4];
  const float* b1  = (const float*)d_in[5];
  const float* g1  = (const float*)d_in[6];
  const float* be1 = (const float*)d_in[7];
  const float* W2  = (const float*)d_in[8];
  const float* as2 = (const float*)d_in[9];
  const float* ad2 = (const float*)d_in[10];
  const float* b2  = (const float*)d_in[11];
  const float* g2  = (const float*)d_in[12];
  const float* be2 = (const float*)d_in[13];
  const float* W3  = (const float*)d_in[14];
  const float* as3 = (const float*)d_in[15];
  const float* ad3 = (const float*)d_in[16];
  const float* b3  = (const float*)d_in[17];
  const float* Wl  = (const float*)d_in[18];
  const float* bl  = (const float*)d_in[19];
  const float* Wd  = (const float*)d_in[20];
  const float* bd  = (const float*)d_in[21];

  // Workspace layout (~136.5 MB, 32B-aligned sections)
  float* ws = (float*)d_ws;
  Scratch S;
  S.B0  = ws;
  S.B1  = S.B0 + (size_t)N_NODES * 256;
  S.A16 = (_Float16*)(S.B1 + (size_t)N_NODES * 256);
  float* p = (float*)(S.A16 + (size_t)N_NODES * 256);
  S.als   = p; p += N_NODES * 2;
  S.ald   = p; p += N_NODES * 2;
  S.emax  = p; p += N_NODES * 2;
  S.denom = p; p += N_NODES * 2;
  S.musum = p; p += 256;
  S.m2sum = p; p += 256;
  S.mu    = p; p += 256;
  S.rstd  = p; p += 256;
  S.wbuf  = p; p += (size_t)ETOT * 2;
  S.Wp    = (_Float16*)p;

  float* adj_out  = (float*)d_out;            // [E]
  float* node_out = adj_out + N_EDGES;        // [N, 128]

  // Layer 1: GATConv(128 -> 2 heads x 128) + ReLU + BN
  run_gat(x, 128, 2, 128, W1, as1, ad1, b1, ei, S, stream);
  run_bn(S.B1, g1, be1, 256, S, stream);
  // Layer 2: GATConv(256 -> 2 heads x 64) + ReLU + BN
  run_gat(S.B1, 256, 2, 64, W2, as2, ad2, b2, ei, S, stream);
  run_bn(S.B1, g2, be2, 128, S, stream);
  // Layer 3: GATConv(128 -> 2 heads x 64) + ReLU
  run_gat(S.B1, 128, 2, 64, W3, as3, ad3, b3, ei, S, stream);
  relu_kernel<<<ceil_div_i(N_NODES * 128, 256), 256, 0, stream>>>(S.B1, N_NODES * 128);

  // z = h @ Wl + bl  (N x 64) -> B0
  {
    int nact = N_NODES * 128;
    cvt_f16_kernel<<<ceil_div_i(nact, 256), 256, 0, stream>>>(S.B1, S.A16, nact);
    launch_gemm(S.A16, Wl, S.Wp, bl, S.B0, 128, 64, stream);
  }
  // node_recon = z @ Wd + bd  -> d_out[E:]
  {
    int nact = N_NODES * 64;
    cvt_f16_kernel<<<ceil_div_i(nact, 256), 256, 0, stream>>>(S.B0, S.A16, nact);
    launch_gemm(S.A16, Wd, S.Wp, bd, node_out, 64, 128, stream);
  }
  // adj_recon = sigmoid(<z[src], z[dst]>) on the original E edges
  adj_decode_kernel<<<ceil_div_i(N_EDGES, 8), 256, 0, stream>>>(ei, S.B0, adj_out);
}
